// FeedForward_94489281268
// MI455X (gfx1250) — matrix operand order repaired and checked by reading, not compiled, not run on hardware
//
#include <hip/hip_runtime.h>
#include <hip/hip_bf16.h>
#include <math.h>
#include <stdint.h>

// ---------------------------------------------------------------------------
// MoE FFN (top-2 of 8 experts, dense-compute formulation) for gfx1250.
//   gate   : comb[n][e] = softmax over top-2 logits (0 elsewhere)
//   gemm1  : h = gelu(x @ w1[e]^T)          (bf16 WMMA, fp32 accum)
//   gemm2  : out (+)= comb[:,e] * (h @ w2[e]^T)
// Shapes: N=4096 tokens, D=1024, F=4096, E=8.
// Double-buffered LDS, one barrier per k-step. GEMM2 A-tile staged with
// global_load_async_to_lds_b128 (ASYNCcnt) since it is already bf16.
// ---------------------------------------------------------------------------

typedef __bf16 bf16;
typedef __bf16 bf16x4 __attribute__((ext_vector_type(4)));
typedef __bf16 bf16x8 __attribute__((ext_vector_type(8)));
typedef __bf16 v16bf  __attribute__((ext_vector_type(16)));
typedef float  v8f    __attribute__((ext_vector_type(8)));

#define MOE_D 1024
#define MOE_F 4096
#define MOE_E 8

#define BM 128
#define BN 128
#define BK 32
#define LDSK 40   // padded LDS row length in bf16 (80 B, keeps 16B chunk alignment)

#define WAIT_ASYNC_0() asm volatile("s_wait_asynccnt 0x0" ::: "memory")

// ---------------------------------------------------------------------------
// Branchless GELU (exact form via Abramowitz-Stegun erf, |err| < 1.5e-7).
// Avoids the exec-mask-divergent OCML erff expansion in the epilogue.
// ---------------------------------------------------------------------------
__device__ __forceinline__ float gelu_f(float v) {
  const float ax = fabsf(v) * 0.70710678118654752f;   // |v|/sqrt(2)
  const float t  = __builtin_amdgcn_rcpf(1.0f + 0.3275911f * ax);
  float p = 1.061405429f;
  p = p * t - 1.453152027f;
  p = p * t + 1.421413741f;
  p = p * t - 0.284496736f;
  p = p * t + 0.254829592f;
  const float er = 1.0f - (p * t) * __expf(-ax * ax);  // erf(|v|/sqrt2)
  const float se = copysignf(er, v);
  return 0.5f * v * (1.0f + se);
}

// ---------------------------------------------------------------------------
// Gating: one block per token, 128 threads. 8 dot products of length D,
// then top-2 + softmax into comb[n][0..7].
// ---------------------------------------------------------------------------
__global__ __launch_bounds__(128)
void moe_gate(const float* __restrict__ x, const float* __restrict__ gw,
              float* __restrict__ comb) {
  const int n = blockIdx.x;
  const int t = threadIdx.x;
  const float* xr = x + (size_t)n * MOE_D;

  float p[MOE_E];
#pragma unroll
  for (int e = 0; e < MOE_E; ++e) p[e] = 0.f;

  for (int d = t; d < MOE_D; d += 128) {
    const float xv = xr[d];
#pragma unroll
    for (int e = 0; e < MOE_E; ++e) p[e] += xv * gw[e * MOE_D + d];
  }

  __shared__ float red[MOE_E][128];
#pragma unroll
  for (int e = 0; e < MOE_E; ++e) red[e][t] = p[e];
  __syncthreads();
  for (int s = 64; s > 0; s >>= 1) {
    if (t < s) {
#pragma unroll
      for (int e = 0; e < MOE_E; ++e) red[e][t] += red[e][t + s];
    }
    __syncthreads();
  }

  if (t == 0) {
    float lg[MOE_E];
#pragma unroll
    for (int e = 0; e < MOE_E; ++e) lg[e] = red[e][0];
    int i0 = 0;
    for (int e = 1; e < MOE_E; ++e) if (lg[e] > lg[i0]) i0 = e;
    int i1 = -1; float b1 = -3.4e38f;
    for (int e = 0; e < MOE_E; ++e) if (e != i0 && lg[e] > b1) { b1 = lg[e]; i1 = e; }
    const float m  = fmaxf(lg[i0], b1);
    const float e0 = __expf(lg[i0] - m);
    const float e1 = __expf(b1 - m);
    const float inv = 1.f / (e0 + e1);
    float w[MOE_E];
#pragma unroll
    for (int e = 0; e < MOE_E; ++e) w[e] = 0.f;
    w[i0] = e0 * inv; w[i1] = e1 * inv;
#pragma unroll
    for (int e = 0; e < MOE_E; ++e) comb[(size_t)n * MOE_E + e] = w[e];
  }
}

// ---------------------------------------------------------------------------
// WMMA fragment loads from LDS (row-major tiles, row stride LDSK bf16).
// A 16x32 bf16 (ISA 7.12.2): lane L -> row M = L&15; VGPR0-3: K = base..base+7,
// VGPR4-7: K = base+16..base+23, base = (L<16)?0:8  -> two 16B chunks.
// ---------------------------------------------------------------------------
__device__ __forceinline__ v16bf load_afrag(const bf16* base, int lane) {
  const int r  = lane & 15;
  const int kh = (lane >> 4) << 3;          // 0 or 8
  const bf16* p = base + r * LDSK + kh;
  v16bf f;
  *(bf16x8*)&f        = *(const bf16x8*)(p);
  *((bf16x8*)&f + 1)  = *(const bf16x8*)(p + 16);
  return f;
}

// B 32x16 bf16 (from row-major W[n][k]): lane L -> column N = L&15,
// K = base..base+7 then base+16..base+23, base = (L<16)?0:8 (matches A).
__device__ __forceinline__ v16bf load_bfrag(const bf16* base, int lane) {
  const int n  = lane & 15;
  const int kh = (lane >> 4) << 3;          // 0 or 8
  const bf16* p = base + n * LDSK + kh;
  v16bf f;
  *(bf16x8*)&f        = *(const bf16x8*)(p);
  *((bf16x8*)&f + 1)  = *(const bf16x8*)(p + 16);
  return f;
}

// ---------------------------------------------------------------------------
// GEMM1: h[N,F] = gelu(x[N,D] @ w1e[F,D]^T), bf16 out.
// 128x128 tile, 8 waves (2x4), wave sub-tile 64x32 = 4x2 WMMA accum tiles.
// Double-buffered LDS, fp32 -> bf16 conversion during staging.
// ---------------------------------------------------------------------------
__global__ __launch_bounds__(256)
void moe_gemm1(const float* __restrict__ x, const float* __restrict__ w1e,
               bf16* __restrict__ h) {
  __shared__ __align__(16) bf16 As[2][BM * LDSK];
  __shared__ __align__(16) bf16 Bs[2][BN * LDSK];

  const int tid  = threadIdx.x;
  const int wave = tid >> 5;
  const int lane = tid & 31;
  const int wm   = wave >> 2;     // 0..1
  const int wn   = wave & 3;      // 0..3
  const int m0   = blockIdx.x * BM;
  const int n0   = blockIdx.y * BN;

  const int sr = tid >> 3;        // staging row base (0..31), +i*32
  const int sc = (tid & 7) * 4;   // staging col (floats)

  v8f acc[4][2];
#pragma unroll
  for (int mi = 0; mi < 4; ++mi)
#pragma unroll
    for (int ni = 0; ni < 2; ++ni)
      acc[mi][ni] = (v8f){0.f, 0.f, 0.f, 0.f, 0.f, 0.f, 0.f, 0.f};

  float4 ga[4], gb[4];

  auto gload = [&](int k0) {
#pragma unroll
    for (int i = 0; i < 4; ++i) {
      const int r = sr + i * 32;
      ga[i] = *(const float4*)(x   + (size_t)(m0 + r) * MOE_D + k0 + sc);
      gb[i] = *(const float4*)(w1e + (size_t)(n0 + r) * MOE_D + k0 + sc);
    }
  };
  auto lstore = [&](int buf) {
#pragma unroll
    for (int i = 0; i < 4; ++i) {
      const int r = sr + i * 32;
      bf16x4 a = {(bf16)ga[i].x, (bf16)ga[i].y, (bf16)ga[i].z, (bf16)ga[i].w};
      bf16x4 b = {(bf16)gb[i].x, (bf16)gb[i].y, (bf16)gb[i].z, (bf16)gb[i].w};
      *(bf16x4*)&As[buf][r * LDSK + sc] = a;
      *(bf16x4*)&Bs[buf][r * LDSK + sc] = b;
    }
  };
  auto compute = [&](int buf) {
    v16bf af[4], bfr[2];
#pragma unroll
    for (int mi = 0; mi < 4; ++mi)
      af[mi] = load_afrag(&As[buf][(wm * 64 + mi * 16) * LDSK], lane);
#pragma unroll
    for (int ni = 0; ni < 2; ++ni)
      bfr[ni] = load_bfrag(&Bs[buf][(wn * 32 + ni * 16) * LDSK], lane);
#pragma unroll
    for (int mi = 0; mi < 4; ++mi)
#pragma unroll
      for (int ni = 0; ni < 2; ++ni)
        acc[mi][ni] = __builtin_amdgcn_wmma_f32_16x16x32_bf16(
            false, af[mi], false, bfr[ni], (short)0, acc[mi][ni], false, false);
  };

  constexpr int NK = MOE_D / BK;  // 32
  gload(0);
  lstore(0);
  gload(BK);
  __syncthreads();

  for (int k = 0; k < NK; ++k) {
    if (k + 1 < NK) {
      lstore((k + 1) & 1);                 // other buffer: readers done last barrier
      if (k + 2 < NK) gload((k + 2) * BK); // prefetch overlaps compute below
    }
    compute(k & 1);
    __syncthreads();
  }

  // Epilogue: branchless GELU, store bf16.  C/D: m = r + 8*(lane>=16), n = lane&15.
#pragma unroll
  for (int mi = 0; mi < 4; ++mi) {
    const int mbase = m0 + wm * 64 + mi * 16 + ((lane >> 4) << 3);
#pragma unroll
    for (int ni = 0; ni < 2; ++ni) {
      const int col = n0 + wn * 32 + ni * 16 + (lane & 15);
#pragma unroll
      for (int r = 0; r < 8; ++r) {
        h[(size_t)(mbase + r) * MOE_F + col] = (bf16)gelu_f(acc[mi][ni][r]);
      }
    }
  }
}

// ---------------------------------------------------------------------------
// GEMM2: out[N,D] (+)= comb[:,e] * (h[N,F] @ w2e[D,F]^T).
// A (h) is bf16 already -> staged via global_load_async_to_lds_b128 (ASYNCcnt).
// B (w2e) is fp32 -> VGPR convert + ds_store path. Double-buffered.
// ---------------------------------------------------------------------------
__global__ __launch_bounds__(256)
void moe_gemm2(const bf16* __restrict__ hsrc, const float* __restrict__ w2e,
               const float* __restrict__ comb, float* __restrict__ out,
               int expert, int init) {
  __shared__ __align__(16) bf16 As[2][BM * LDSK];
  __shared__ __align__(16) bf16 Bs[2][BN * LDSK];

  const int tid  = threadIdx.x;
  const int wave = tid >> 5;
  const int lane = tid & 31;
  const int wm   = wave >> 2;
  const int wn   = wave & 3;
  const int m0   = blockIdx.x * BM;
  const int n0   = blockIdx.y * BN;

  const int sr = tid >> 3;        // B staging row base, +i*32
  const int sc = (tid & 7) * 4;   // B staging col (floats)
  const int ar = tid >> 2;        // A staging row base, +i*64
  const int ac = (tid & 3) * 8;   // A staging col (bf16, 16B chunk)

  v8f acc[4][2];
#pragma unroll
  for (int mi = 0; mi < 4; ++mi)
#pragma unroll
    for (int ni = 0; ni < 2; ++ni)
      acc[mi][ni] = (v8f){0.f, 0.f, 0.f, 0.f, 0.f, 0.f, 0.f, 0.f};

  float4 gb[4];

  auto a_async = [&](int k0, int buf) {
#pragma unroll
    for (int i = 0; i < 2; ++i) {
      const int r = ar + i * 64;
      const bf16* gp = hsrc + (size_t)(m0 + r) * MOE_F + k0 + ac;
      const unsigned loff = (unsigned)(uintptr_t)&As[buf][r * LDSK + ac];
      asm volatile("global_load_async_to_lds_b128 %0, %1, off"
                   :: "v"(loff), "v"(gp) : "memory");
    }
  };
  auto b_gload = [&](int k0) {
#pragma unroll
    for (int i = 0; i < 4; ++i) {
      const int r = sr + i * 32;
      gb[i] = *(const float4*)(w2e + (size_t)(n0 + r) * MOE_F + k0 + sc);
    }
  };
  auto b_lstore = [&](int buf) {
#pragma unroll
    for (int i = 0; i < 4; ++i) {
      const int r = sr + i * 32;
      bf16x4 b = {(bf16)gb[i].x, (bf16)gb[i].y, (bf16)gb[i].z, (bf16)gb[i].w};
      *(bf16x4*)&Bs[buf][r * LDSK + sc] = b;
    }
  };
  auto compute = [&](int buf) {
    v16bf af[4], bfr[2];
#pragma unroll
    for (int mi = 0; mi < 4; ++mi)
      af[mi] = load_afrag(&As[buf][(wm * 64 + mi * 16) * LDSK], lane);
#pragma unroll
    for (int ni = 0; ni < 2; ++ni)
      bfr[ni] = load_bfrag(&Bs[buf][(wn * 32 + ni * 16) * LDSK], lane);
#pragma unroll
    for (int mi = 0; mi < 4; ++mi)
#pragma unroll
      for (int ni = 0; ni < 2; ++ni)
        acc[mi][ni] = __builtin_amdgcn_wmma_f32_16x16x32_bf16(
            false, af[mi], false, bfr[ni], (short)0, acc[mi][ni], false, false);
  };

  constexpr int NK = MOE_F / BK;  // 128
  b_gload(0);
  a_async(0, 0);
  b_lstore(0);
  b_gload(BK);
  WAIT_ASYNC_0();
  __syncthreads();

  for (int k = 0; k < NK; ++k) {
    if (k + 1 < NK) {
      a_async((k + 1) * BK, (k + 1) & 1);   // async fill of other buffer
      b_lstore((k + 1) & 1);
      if (k + 2 < NK) b_gload((k + 2) * BK);
    }
    compute(k & 1);                          // hides async latency
    if (k + 1 < NK) WAIT_ASYNC_0();          // drain before barrier
    __syncthreads();
  }

  // Epilogue: scale by per-token combine weight; init overwrites poisoned out.
#pragma unroll
  for (int mi = 0; mi < 4; ++mi) {
    const int mbase = m0 + wm * 64 + mi * 16 + ((lane >> 4) << 3);
    float cw[8];
#pragma unroll
    for (int r = 0; r < 8; ++r)
      cw[r] = comb[(size_t)(mbase + r) * MOE_E + expert];
#pragma unroll
    for (int ni = 0; ni < 2; ++ni) {
      const int col = n0 + wn * 32 + ni * 16 + (lane & 15);
#pragma unroll
      for (int r = 0; r < 8; ++r) {
        const float v = cw[r] * acc[mi][ni][r];
        const size_t o = (size_t)(mbase + r) * MOE_D + col;
        out[o] = init ? v : (out[o] + v);
      }
    }
  }
}

// ---------------------------------------------------------------------------
extern "C" void kernel_launch(void* const* d_in, const int* in_sizes, int n_in,
                              void* d_out, int out_size, void* d_ws, size_t ws_size,
                              hipStream_t stream) {
  const float* x      = (const float*)d_in[0];   // [B,S,D]
  const float* gate_w = (const float*)d_in[1];   // [E,D]
  const float* w1     = (const float*)d_in[2];   // [E,F,D]
  const float* w2     = (const float*)d_in[3];   // [E,D,F]
  float* out          = (float*)d_out;           // [B,S,D]

  const int N = in_sizes[0] / MOE_D;             // B*S = 4096

  // Workspace: comb[N][8] fp32, then h[N][F] bf16 (reused per expert).
  float* comb = (float*)d_ws;
  size_t off  = ((size_t)N * MOE_E * sizeof(float) + 255) & ~(size_t)255;
  bf16*  h    = (bf16*)((char*)d_ws + off);

  moe_gate<<<N, 128, 0, stream>>>(x, gate_w, comb);

  dim3 g1(N / BM, MOE_F / BN);
  dim3 g2(N / BM, MOE_D / BN);
  for (int e = 0; e < MOE_E; ++e) {
    moe_gemm1<<<g1, 256, 0, stream>>>(x, w1 + (size_t)e * MOE_F * MOE_D, h);
    moe_gemm2<<<g2, 256, 0, stream>>>(h, w2 + (size_t)e * MOE_D * MOE_F,
                                      comb, out, e, e == 0 ? 1 : 0);
  }
}
